// GPT_62362925138694
// MI455X (gfx1250) — matrix operand order repaired and checked by reading, not compiled, not run on hardware
//
#include <hip/hip_runtime.h>

// ---------------------------------------------------------------------------
// GPT forward (B=4,T=1024,C=768,L=6,H=12,HD=64,V=32000) for gfx1250 (MI455X)
// f16 WMMA (V_WMMA_F32_16X16X32_F16) with f32 accumulate everywhere.
// ---------------------------------------------------------------------------

typedef _Float16 h16;
typedef __attribute__((ext_vector_type(16))) _Float16 v16h;
typedef __attribute__((ext_vector_type(8)))  float    v8f;

#define GB  4
#define GT  1024
#define GV  32000
#define GC  768
#define GL  6
#define GH  12
#define GHD 64
#define GF  3072
#define GBT (GB * GT)

// ---------------------------------------------------------------------------
// Fragment loaders (contiguous K order, used by this code for both A and B):
//
// A fragment (16x32 f16, 8 VGPRs): each lane holds half of one row of A
// -- lanes 0-15 hold M=lane with K=0..15; lanes 16-31 hold
// M=lane-16 with K=16..31.
//
// B fragment (32x16 f16, 8 VGPRs): each lane holds half of one column of B
// -- lane = N; the 16 halves per lane are contiguous K=0..15 for lanes
// 0-15 and K=16..31 for lanes 16-31 (matches the sparse-B table pattern).
// ---------------------------------------------------------------------------
__device__ inline v16h load_frag_a(const h16* __restrict__ tile, int ld, int k0) {
  int lane = threadIdx.x & 31;
  const h16* p = tile + (size_t)(lane & 15) * ld + k0 + ((lane >> 4) << 4);
  v16h f;
#pragma unroll
  for (int r = 0; r < 8; ++r) {
    int koff = 2 * r;  // 0,2,4,...,14
    f[2 * r]     = p[koff];
    f[2 * r + 1] = p[koff + 1];
  }
  return f;
}

// B fragment from an N-major [N,K] matrix (row stride ld): per lane this is
// 16 contiguous halves (32 bytes) -> two global_load_b128.
__device__ inline v16h load_frag_b(const h16* __restrict__ tile, int ld, int k0) {
  int lane = threadIdx.x & 31;
  const h16* p = tile + (size_t)(lane & 15) * ld + k0 + ((lane >> 4) << 4);
  return *(const v16h*)p;
}

// B fragment gathered from a K-major [K,N] matrix (row stride ld), used for V
// in attention (V is stored [t, d] and we need d as the N dimension).
__device__ inline v16h load_frag_b_kmaj(const h16* __restrict__ tile, int ld, int k0) {
  int lane = threadIdx.x & 31;
  int n  = lane & 15;
  int kb = k0 + ((lane >> 4) << 4);
  v16h f;
#pragma unroll
  for (int e = 0; e < 16; ++e)
    f[e] = tile[(size_t)(kb + e) * ld + n];
  return f;
}

// ---------------------------------------------------------------------------
// Weight cast + transpose: src f32 [batch, rows, cols] -> dst f16 [batch, cols, rows]
// ---------------------------------------------------------------------------
__global__ __launch_bounds__(256) void transpose_cast_kernel(
    const float* __restrict__ src, h16* __restrict__ dst,
    int rows, int cols, long total) {
  long g = (long)blockIdx.x * 256 + threadIdx.x;
  if (g >= total) return;
  long rc  = (long)rows * cols;
  long b   = g / rc;
  int  rem = (int)(g - b * rc);
  int  r   = rem / cols;
  int  c   = rem - r * cols;
  dst[b * rc + (long)c * rows + r] = (h16)src[g];
}

// ---------------------------------------------------------------------------
// x = tok_emb[idx] + pos_emb
// ---------------------------------------------------------------------------
__global__ __launch_bounds__(256) void embed_kernel(
    const int* __restrict__ idx, const float* __restrict__ tok,
    const float* __restrict__ pos, float* __restrict__ x) {
  long g = (long)blockIdx.x * 256 + threadIdx.x;
  if (g >= (long)GBT * GC) return;
  int c  = (int)(g % GC);
  int bt = (int)(g / GC);
  int t  = bt % GT;
  x[g] = tok[(size_t)idx[bt] * GC + c] + pos[(size_t)t * GC + c];
}

// ---------------------------------------------------------------------------
// LayerNorm: one block per row, f16 output for the downstream WMMA GEMMs.
// ---------------------------------------------------------------------------
__global__ __launch_bounds__(256) void layernorm_kernel(
    const float* __restrict__ x, const float* __restrict__ s,
    const float* __restrict__ bi, h16* __restrict__ out, int Cc) {
  __shared__ float buf[256];
  int row = blockIdx.x;
  int tid = threadIdx.x;
  const float* xr = x + (size_t)row * Cc;

  float sum = 0.f;
  for (int c = tid; c < Cc; c += 256) sum += xr[c];
  buf[tid] = sum; __syncthreads();
  for (int st = 128; st > 0; st >>= 1) {
    if (tid < st) buf[tid] += buf[tid + st];
    __syncthreads();
  }
  float mu = buf[0] / Cc;
  __syncthreads();

  float vs = 0.f;
  for (int c = tid; c < Cc; c += 256) { float d = xr[c] - mu; vs += d * d; }
  buf[tid] = vs; __syncthreads();
  for (int st = 128; st > 0; st >>= 1) {
    if (tid < st) buf[tid] += buf[tid + st];
    __syncthreads();
  }
  float inv = rsqrtf(buf[0] / Cc + 1e-5f);

  for (int c = tid; c < Cc; c += 256)
    out[(size_t)row * Cc + c] = (h16)((xr[c] - mu) * inv * s[c] + bi[c]);
}

// ---------------------------------------------------------------------------
// Generic WMMA GEMM: Out[M,N] = A[M,K](f16) x Bt[N,K](f16)^T (+bias)(+relu)
// Block = 8 waves, block tile 32(M) x 128(N). Each wave: one B fragment per
// k-step reused across two M tiles (2 accumulators).
// grid = (N/128, M/32). All problem dims divide the tiles exactly.
// ---------------------------------------------------------------------------
template <bool OUT_HALF, bool RELU, bool RESID, bool BIAS>
__global__ __launch_bounds__(256) void gemm_wmma_kernel(
    const h16* __restrict__ A, const h16* __restrict__ Bt,
    const float* __restrict__ bias, float* __restrict__ outF,
    h16* __restrict__ outH, int K, int lda, int ldb, int ldo) {
  int wave = threadIdx.x >> 5;
  int lane = threadIdx.x & 31;
  int m0 = blockIdx.y << 5;                       // 32 rows / block
  int n0 = (blockIdx.x << 7) + (wave << 4);       // 16 cols / wave

  const h16* at0 = A + (size_t)m0 * lda;
  const h16* at1 = at0 + (size_t)16 * lda;
  const h16* bt  = Bt + (size_t)n0 * ldb;

  v8f acc0 = {};
  v8f acc1 = {};
  for (int k0 = 0; k0 < K; k0 += 32) {
    v16h b  = load_frag_b(bt,  ldb, k0);
    v16h a0 = load_frag_a(at0, lda, k0);
    v16h a1 = load_frag_a(at1, lda, k0);
    if (k0 + 32 < K)  // stage next B k-slab (global_prefetch_b8)
      __builtin_prefetch(bt + (size_t)(lane & 15) * ldb + k0 + 32, 0, 1);
    acc0 = __builtin_amdgcn_wmma_f32_16x16x32_f16(false, a0, false, b, (short)0, acc0, false, false);
    acc1 = __builtin_amdgcn_wmma_f32_16x16x32_f16(false, a1, false, b, (short)0, acc1, false, false);
  }

  int n  = n0 + (lane & 15);
  int mb = m0 + ((lane >> 4) << 3);               // C/D: M = r + (hi?8:0)
  float bv = BIAS ? bias[n] : 0.f;
#pragma unroll
  for (int r = 0; r < 8; ++r) {
#pragma unroll
    for (int half = 0; half < 2; ++half) {
      int m = mb + r + half * 16;
      float val = (half ? acc1[r] : acc0[r]) + bv;
      if (RELU) val = fmaxf(val, 0.f);
      size_t o = (size_t)m * ldo + n;
      if (OUT_HALF)      outH[o] = (h16)val;
      else if (RESID)    outF[o] += val;
      else               outF[o] = val;
    }
  }
}

// ---------------------------------------------------------------------------
// Fused causal attention for one (b, h, 16-query tile).
// Phase 1: S = Q Kt / sqrt(HD) via WMMA -> f16 LDS [16 x 1024] (masked).
// Phase 2: row softmax (unnormalized p kept in LDS, row sums in LDS).
// Phase 3: P x V via WMMA, normalization in epilogue, residual-add into x.
// grid = (T/16, H, B), block = 256 (8 waves).
// ---------------------------------------------------------------------------
__global__ __launch_bounds__(256) void attention_kernel(
    const h16* __restrict__ q, const h16* __restrict__ k,
    const h16* __restrict__ v, float* __restrict__ x) {
  __shared__ h16  sc[16 * GT];   // 32 KB score / prob buffer
  __shared__ float red[256];
  __shared__ float rmax[16];
  __shared__ float rs[16];

  int m0   = blockIdx.x << 4;
  int hh   = blockIdx.y;
  int bb   = blockIdx.z;
  int tid  = threadIdx.x;
  int wave = tid >> 5;
  int lane = tid & 31;

  const h16* qb = q + ((size_t)bb * GT + m0) * GC + hh * GHD;
  const h16* kb = k + (size_t)bb * GT * GC + hh * GHD;

  // Phase 1: scores (A = Q rows, B = K rows as N-major [t, d])
  v16h aq0 = load_frag_a(qb, GC, 0);
  v16h aq1 = load_frag_a(qb, GC, 32);
  for (int nt = wave; nt < GT / 16; nt += 8) {
    int nb = nt << 4;
    const h16* kt = kb + (size_t)nb * GC;
    v16h b0 = load_frag_b(kt, GC, 0);
    v16h b1 = load_frag_b(kt, GC, 32);
    v8f acc = {};
    acc = __builtin_amdgcn_wmma_f32_16x16x32_f16(false, aq0, false, b0, (short)0, acc, false, false);
    acc = __builtin_amdgcn_wmma_f32_16x16x32_f16(false, aq1, false, b1, (short)0, acc, false, false);
    int n  = nb + (lane & 15);
    int mb = (lane >> 4) << 3;
#pragma unroll
    for (int r = 0; r < 8; ++r) {
      int mi = mb + r;
      float sv = acc[r] * 0.125f;                 // HD^-0.5
      if (n > m0 + mi) sv = -65504.f;             // causal mask (finite for f16)
      sc[mi * GT + n] = (h16)sv;
    }
  }
  __syncthreads();

  // Phase 2: softmax (16 threads per row)
  int row = tid >> 4, sub = tid & 15;
  float mx = -65504.f;
  for (int c = sub; c < GT; c += 16) mx = fmaxf(mx, (float)sc[row * GT + c]);
  red[tid] = mx; __syncthreads();
  if (sub == 0) {
    float m2 = -65504.f;
    for (int i = 0; i < 16; ++i) m2 = fmaxf(m2, red[(row << 4) + i]);
    rmax[row] = m2;
  }
  __syncthreads();
  mx = rmax[row];
  float sum = 0.f;
  for (int c = sub; c < GT; c += 16) {
    float p = __expf((float)sc[row * GT + c] - mx);
    sc[row * GT + c] = (h16)p;
    sum += p;
  }
  red[tid] = sum; __syncthreads();
  if (sub == 0) {
    float s2 = 0.f;
    for (int i = 0; i < 16; ++i) s2 += red[(row << 4) + i];
    rs[row] = s2;
  }
  __syncthreads();

  // Phase 3: att = P x V (waves 0..3, one 16x16 tile of the 16x64 output each)
  if (wave < 4) {
    int n0 = wave << 4;
    const h16* vb = v + (size_t)bb * GT * GC + hh * GHD + n0;
    v8f acc = {};
    for (int k0 = 0; k0 < GT; k0 += 32) {
      v16h aw = load_frag_a(sc, GT, k0);
      v16h bv = load_frag_b_kmaj(vb, GC, k0);
      acc = __builtin_amdgcn_wmma_f32_16x16x32_f16(false, aw, false, bv, (short)0, acc, false, false);
    }
    int nl = n0 + (lane & 15);
    int mb = (lane >> 4) << 3;
#pragma unroll
    for (int r = 0; r < 8; ++r) {
      int mi = mb + r;
      float invs = 1.f / rs[mi];
      size_t o = ((size_t)bb * GT + m0 + mi) * GC + hh * GHD + nl;
      x[o] += acc[r] * invs;                      // residual add (disjoint)
    }
  }
}

// ---------------------------------------------------------------------------
// Loss: per-row -(logit[target] - logsumexp), then deterministic final mean.
// ---------------------------------------------------------------------------
__global__ __launch_bounds__(256) void loss_rows_kernel(
    const float* __restrict__ logits, const int* __restrict__ targets,
    float* __restrict__ rowloss) {
  __shared__ float buf[256];
  int row = blockIdx.x;
  int tid = threadIdx.x;
  const float* lr = logits + (size_t)row * GV;

  float mx = -3.4e38f;
  for (int c = tid; c < GV; c += 256) mx = fmaxf(mx, lr[c]);
  buf[tid] = mx; __syncthreads();
  for (int st = 128; st > 0; st >>= 1) {
    if (tid < st) buf[tid] = fmaxf(buf[tid], buf[tid + st]);
    __syncthreads();
  }
  mx = buf[0];
  __syncthreads();

  float s = 0.f;
  for (int c = tid; c < GV; c += 256) s += expf(lr[c] - mx);
  buf[tid] = s; __syncthreads();
  for (int st = 128; st > 0; st >>= 1) {
    if (tid < st) buf[tid] += buf[tid + st];
    __syncthreads();
  }
  if (tid == 0) {
    float lse = mx + logf(buf[0]);
    rowloss[row] = -(lr[targets[row]] - lse);
  }
}

__global__ __launch_bounds__(256) void loss_finalize_kernel(
    const float* __restrict__ rowloss, float* __restrict__ out) {
  __shared__ float buf[256];
  int tid = threadIdx.x;
  float s = 0.f;
  for (int i = tid; i < GBT; i += 256) s += rowloss[i];
  buf[tid] = s; __syncthreads();
  for (int st = 128; st > 0; st >>= 1) {
    if (tid < st) buf[tid] += buf[tid + st];
    __syncthreads();
  }
  if (tid == 0) out[0] = buf[0] / (float)GBT;
}

// ---------------------------------------------------------------------------
// Host launcher
// ---------------------------------------------------------------------------
extern "C" void kernel_launch(void* const* d_in, const int* in_sizes, int n_in,
                              void* d_out, int out_size, void* d_ws, size_t ws_size,
                              hipStream_t stream) {
  (void)in_sizes; (void)n_in; (void)out_size; (void)ws_size;

  const int*   idx     = (const int*)  d_in[0];
  const int*   targets = (const int*)  d_in[1];
  const float* tok_emb = (const float*)d_in[2];
  const float* pos_emb = (const float*)d_in[3];
  const float* Wq      = (const float*)d_in[4];
  const float* Wk      = (const float*)d_in[5];
  const float* Wv      = (const float*)d_in[6];
  const float* ln1_s   = (const float*)d_in[7];
  const float* ln1_b   = (const float*)d_in[8];
  const float* ln2_s   = (const float*)d_in[9];
  const float* ln2_b   = (const float*)d_in[10];
  const float* W1      = (const float*)d_in[11];
  const float* b1      = (const float*)d_in[12];
  const float* W2      = (const float*)d_in[13];
  const float* b2      = (const float*)d_in[14];
  const float* lnf_s   = (const float*)d_in[15];
  const float* lnf_b   = (const float*)d_in[16];
  const float* Wout    = (const float*)d_in[17];
  const float* bout    = (const float*)d_in[18];

  float* logits = (float*)d_out;
  float* loss   = logits + (size_t)GBT * GV;

  // Workspace layout (256B-aligned bump allocator)
  char* ws = (char*)d_ws;
  size_t off = 0;
  auto alloc = [&](size_t bytes) -> void* {
    void* p = ws + off;
    off = (off + bytes + 255) & ~(size_t)255;
    return p;
  };

  float* x    = (float*)alloc((size_t)GBT * GC * 4);
  h16* xn16   = (h16*)  alloc((size_t)GBT * GC * 2);
  h16* q16    = (h16*)  alloc((size_t)GBT * GC * 2);
  h16* k16    = (h16*)  alloc((size_t)GBT * GC * 2);
  h16* v16    = (h16*)  alloc((size_t)GBT * GC * 2);
  h16* hid16  = (h16*)  alloc((size_t)GBT * GF * 2);
  float* rowl = (float*)alloc((size_t)GBT * 4);
  h16* WqT    = (h16*)  alloc((size_t)GL * GC * GC * 2);   // [L,H,HD,C]
  h16* WkT    = (h16*)  alloc((size_t)GL * GC * GC * 2);
  h16* WvT    = (h16*)  alloc((size_t)GL * GC * GC * 2);
  h16* W1T    = (h16*)  alloc((size_t)GL * GF * GC * 2);   // [L,4C,C]
  h16* W2T    = (h16*)  alloc((size_t)GL * GC * GF * 2);   // [L,C,4C]
  h16* WoutT  = (h16*)  alloc((size_t)GV * GC * 2);        // [V,C]

  auto blocks = [](long total) { return (unsigned)((total + 255) / 256); };

  // --- weight cast+transpose to f16 N-major layouts -------------------------
  {
    long n = (long)GL * GH * GC * GHD;     // Wq/Wk/Wv: batch L*H, [C,HD]->[HD,C]
    transpose_cast_kernel<<<blocks(n), 256, 0, stream>>>(Wq, WqT, GC, GHD, n);
    transpose_cast_kernel<<<blocks(n), 256, 0, stream>>>(Wk, WkT, GC, GHD, n);
    transpose_cast_kernel<<<blocks(n), 256, 0, stream>>>(Wv, WvT, GC, GHD, n);
    long n1 = (long)GL * GC * GF;          // W1: batch L, [C,4C]->[4C,C]
    transpose_cast_kernel<<<blocks(n1), 256, 0, stream>>>(W1, W1T, GC, GF, n1);
    long n2 = (long)GL * GF * GC;          // W2: batch L, [4C,C]->[C,4C]
    transpose_cast_kernel<<<blocks(n2), 256, 0, stream>>>(W2, W2T, GF, GC, n2);
    long no = (long)GC * GV;               // Wout: [C,V]->[V,C]
    transpose_cast_kernel<<<blocks(no), 256, 0, stream>>>(Wout, WoutT, GC, GV, no);
  }

  // --- embedding ------------------------------------------------------------
  embed_kernel<<<blocks((long)GBT * GC), 256, 0, stream>>>(idx, tok_emb, pos_emb, x);

  // --- transformer layers ---------------------------------------------------
  for (int l = 0; l < GL; ++l) {
    layernorm_kernel<<<GBT, 256, 0, stream>>>(x, ln1_s + l * GC, ln1_b + l * GC, xn16, GC);

    dim3 gqkv(GC / 128, GBT / 32);
    gemm_wmma_kernel<true, false, false, false><<<gqkv, 256, 0, stream>>>(
        xn16, WqT + (size_t)l * GC * GC, nullptr, nullptr, q16, GC, GC, GC, GC);
    gemm_wmma_kernel<true, false, false, false><<<gqkv, 256, 0, stream>>>(
        xn16, WkT + (size_t)l * GC * GC, nullptr, nullptr, k16, GC, GC, GC, GC);
    gemm_wmma_kernel<true, false, false, false><<<gqkv, 256, 0, stream>>>(
        xn16, WvT + (size_t)l * GC * GC, nullptr, nullptr, v16, GC, GC, GC, GC);

    attention_kernel<<<dim3(GT / 16, GH, GB), 256, 0, stream>>>(q16, k16, v16, x);

    layernorm_kernel<<<GBT, 256, 0, stream>>>(x, ln2_s + l * GC, ln2_b + l * GC, xn16, GC);

    gemm_wmma_kernel<true, true, false, true><<<dim3(GF / 128, GBT / 32), 256, 0, stream>>>(
        xn16, W1T + (size_t)l * GF * GC, b1 + (size_t)l * GF, nullptr, hid16,
        GC, GC, GC, GF);

    gemm_wmma_kernel<false, false, true, true><<<dim3(GC / 128, GBT / 32), 256, 0, stream>>>(
        hid16, W2T + (size_t)l * GC * GF, b2 + (size_t)l * GC, x, nullptr,
        GF, GF, GF, GC);
  }

  // --- final LN + logits + loss --------------------------------------------
  layernorm_kernel<<<GBT, 256, 0, stream>>>(x, lnf_s, lnf_b, xn16, GC);

  gemm_wmma_kernel<false, false, false, true><<<dim3(GV / 128, GBT / 32), 256, 0, stream>>>(
      xn16, WoutT, bout, logits, nullptr, GC, GC, GC, GV);

  loss_rows_kernel<<<GBT, 256, 0, stream>>>(logits, targets, rowl);
  loss_finalize_kernel<<<1, 256, 0, stream>>>(rowl, loss);
}
